// PairwiseMessages_55027120996928
// MI455X (gfx1250) — matrix operand order repaired and checked by reading, not compiled, not run on hardware
//
#include <hip/hip_runtime.h>

#define NQ 512
#define NK 512
#define D_EQUI 64
#define D_QINV 256
#define D_MSG 128
#define D_OUT 128
#define D_FF 256
#define D_EDGE 64
#define IN_FEATS 320   // 128 pairwise + 64 dot + 64 dist + 64 edges
#define W2_PAD 264     // padded row stride (bf16 elems) for bank-conflict-free b128 reads

typedef __bf16  bf16_t;
typedef bf16_t  v16bf __attribute__((ext_vector_type(16)));
typedef float   v8f   __attribute__((ext_vector_type(8)));
typedef float   f32x4 __attribute__((ext_vector_type(4)));
typedef unsigned int u32x4 __attribute__((ext_vector_type(4)));
typedef int     i32x4 __attribute__((ext_vector_type(4)));

typedef __attribute__((address_space(1))) i32x4* glob_v4i_ptr;
typedef __attribute__((address_space(3))) i32x4* lds_v4i_ptr;

union Frag16 {
    v16bf v;
    u32x4 q[2];
    unsigned short u[16];
};

__device__ __forceinline__ unsigned short f2bf(float f) {
    union { float f; unsigned u; } c; c.f = f;
    unsigned u = c.u;
    u += 0x7FFFu + ((u >> 16) & 1u);        // round-to-nearest-even
    return (unsigned short)(u >> 16);
}

// ---------------- prep kernels (negligible cost) ----------------

__global__ void prep_msg(const float* __restrict__ q_inv, const float* __restrict__ k_inv,
                         const float* __restrict__ qW, const float* __restrict__ qb,
                         const float* __restrict__ kW, const float* __restrict__ kb,
                         float* __restrict__ qmsg, float* __restrict__ kmsg) {
    const int row = blockIdx.x;          // 0..1023
    const int n   = threadIdx.x;         // 0..127
    const float* x; const float* W; const float* bia; float* out;
    if (row < NQ) { x = q_inv + row * D_QINV; W = qW; bia = qb; out = qmsg + row * D_MSG; }
    else          { const int r = row - NQ;
                    x = k_inv + r * D_QINV; W = kW; bia = kb; out = kmsg + r * D_MSG; }
    const float* w = W + n * D_QINV;
    float acc = 0.f;
    for (int j = 0; j < D_QINV; j += 4) {
        f32x4 xv = *(const f32x4*)(x + j);
        f32x4 wv = *(const f32x4*)(w + j);
        acc += xv.x * wv.x + xv.y * wv.y + xv.z * wv.z + xv.w * wv.w;
    }
    out[n] = acc + bia[n];
}

__global__ void prep_norms(const float* __restrict__ qe, const float* __restrict__ ke,
                           float* __restrict__ qn2, float* __restrict__ kn2) {
    const int idx   = blockIdx.x * blockDim.x + threadIdx.x;  // 2*512*64
    const int which = idx >> 15;
    const int i     = idx & 32767;
    const int row   = i >> 6, d = i & 63;
    const float* src = which ? ke : qe;
    const float a0 = src[row * 192 + d];
    const float a1 = src[row * 192 + 64 + d];
    const float a2 = src[row * 192 + 128 + d];
    (which ? kn2 : qn2)[i] = a0 * a0 + a1 * a1 + a2 * a2;
}

__global__ void prep_wcvt(const float* __restrict__ W1, const float* __restrict__ W2,
                          unsigned short* __restrict__ w1b, unsigned short* __restrict__ w2b) {
    const int idx = blockIdx.x * blockDim.x + threadIdx.x;
    const int n1  = D_FF * IN_FEATS;                 // 81920
    if (idx < n1) w1b[idx] = f2bf(W1[idx]);
    else {
        const int j = idx - n1;
        if (j < D_OUT * D_FF) w2b[j] = f2bf(W2[j]);
    }
}

// ---------------- main fused kernel ----------------
// block = 256 threads = 8 waves; wave w: q = qb*8+w, k in [kb*16, kb*16+16)
// GEMM1 B streams via vmem/L0; GEMM2 B is staged once into LDS (async) so the
// two WMMA operand streams use independent delivery paths.

__global__ void __launch_bounds__(256)
pairwise_main(const float* __restrict__ q_equi, const float* __restrict__ k_equi,
              const float* __restrict__ edges,
              const float* __restrict__ bias1, const float* __restrict__ bias2,
              const float* __restrict__ qmsg, const float* __restrict__ kmsg,
              const float* __restrict__ qn2, const float* __restrict__ kn2,
              const unsigned short* __restrict__ w1b, const unsigned short* __restrict__ w2b,
              float* __restrict__ out) {
    __shared__ __align__(16) unsigned short w2lds[D_OUT][W2_PAD];  // ~67 KB, padded rows
    __shared__ __align__(16) unsigned short hT[8][16][32];         // per-wave h transpose tile

    const int wave = threadIdx.x >> 5;
    const int lane = threadIdx.x & 31;
    const int m    = lane & 15;          // A row / B-C column-within-tile
    const int h    = (lane >> 4) & 1;    // half-wave

    const int qb = blockIdx.x >> 5;      // 64 q-blocks
    const int kb = blockIdx.x & 31;      // 32 k-blocks
    const int q  = qb * 8 + wave;
    const int k0 = kb * 16;
    const int k  = k0 + m;

    // ---- kick off async staging of W2 (bf16) into LDS; overlap with feats build ----
#if __has_builtin(__builtin_amdgcn_global_load_async_to_lds_b128)
    for (int c = threadIdx.x; c < (D_OUT * D_FF) / 8; c += 256) {
        const int row = c >> 5;            // 32 chunks of 8 bf16 per 256-wide row
        const int col = (c & 31) * 8;
        __builtin_amdgcn_global_load_async_to_lds_b128(
            (glob_v4i_ptr)(w2b + c * 8),
            (lds_v4i_ptr)&w2lds[row][col], 0, 0);
    }
#else
    for (int c = threadIdx.x; c < (D_OUT * D_FF) / 8; c += 256) {
        const int row = c >> 5;
        const int col = (c & 31) * 8;
        *(u32x4*)&w2lds[row][col] = *(const u32x4*)(w2b + c * 8);
    }
#endif

    const float* qm = qmsg + q * D_MSG;
    const float* km = kmsg + k * D_MSG;
    const float* qe = q_equi + q * (3 * D_EQUI);
    const float* ke = k_equi + k * (3 * D_EQUI);
    const float* qn = qn2 + q * D_EQUI;
    const float* kn = kn2 + k * D_EQUI;
    const float* ed = edges + (q * NK + k) * D_EDGE;

    // ---- build the 10 bf16 A-fragments of feats (16x32 each), ISA A layout ----
    Frag16 afr[10];
#pragma unroll
    for (int ks = 0; ks < 10; ++ks) {
#pragma unroll
        for (int g = 0; g < 2; ++g) {
            const int cc = 32 * ks + 16 * g;     // compile-time segment selector
            const int cb = cc + 8 * h;           // first of an 8-column run
            float f[8];
            if (cc < 128) {                      // pairwise q_msg*k_msg
                const f32x4 qa = *(const f32x4*)(qm + cb);
                const f32x4 qd = *(const f32x4*)(qm + cb + 4);
                const f32x4 ka = *(const f32x4*)(km + cb);
                const f32x4 kd = *(const f32x4*)(km + cb + 4);
                f[0] = qa.x * ka.x; f[1] = qa.y * ka.y; f[2] = qa.z * ka.z; f[3] = qa.w * ka.w;
                f[4] = qd.x * kd.x; f[5] = qd.y * kd.y; f[6] = qd.z * kd.z; f[7] = qd.w * kd.w;
            } else if (cc < 192) {               // channel-wise dot over e=0..2
                const int d = cb - 128;
#pragma unroll
                for (int j = 0; j < 8; ++j) f[j] = 0.f;
#pragma unroll
                for (int e = 0; e < 3; ++e) {
                    const f32x4 qa = *(const f32x4*)(qe + e * D_EQUI + d);
                    const f32x4 qd = *(const f32x4*)(qe + e * D_EQUI + d + 4);
                    const f32x4 ka = *(const f32x4*)(ke + e * D_EQUI + d);
                    const f32x4 kd = *(const f32x4*)(ke + e * D_EQUI + d + 4);
                    f[0] += qa.x * ka.x; f[1] += qa.y * ka.y; f[2] += qa.z * ka.z; f[3] += qa.w * ka.w;
                    f[4] += qd.x * kd.x; f[5] += qd.y * kd.y; f[6] += qd.z * kd.z; f[7] += qd.w * kd.w;
                }
            } else if (cc < 256) {               // dist = sqrt(|q|^2+|k|^2-2*dot)
                const int d = cb - 192;
                float dt[8];
#pragma unroll
                for (int j = 0; j < 8; ++j) dt[j] = 0.f;
#pragma unroll
                for (int e = 0; e < 3; ++e) {
                    const f32x4 qa = *(const f32x4*)(qe + e * D_EQUI + d);
                    const f32x4 qd = *(const f32x4*)(qe + e * D_EQUI + d + 4);
                    const f32x4 ka = *(const f32x4*)(ke + e * D_EQUI + d);
                    const f32x4 kd = *(const f32x4*)(ke + e * D_EQUI + d + 4);
                    dt[0] += qa.x * ka.x; dt[1] += qa.y * ka.y; dt[2] += qa.z * ka.z; dt[3] += qa.w * ka.w;
                    dt[4] += qd.x * kd.x; dt[5] += qd.y * kd.y; dt[6] += qd.z * kd.z; dt[7] += qd.w * kd.w;
                }
                const f32x4 na = *(const f32x4*)(qn + d);
                const f32x4 nd = *(const f32x4*)(qn + d + 4);
                const f32x4 ma = *(const f32x4*)(kn + d);
                const f32x4 md = *(const f32x4*)(kn + d + 4);
                f[0] = sqrtf(fmaxf(na.x + ma.x - 2.f * dt[0], 0.f));
                f[1] = sqrtf(fmaxf(na.y + ma.y - 2.f * dt[1], 0.f));
                f[2] = sqrtf(fmaxf(na.z + ma.z - 2.f * dt[2], 0.f));
                f[3] = sqrtf(fmaxf(na.w + ma.w - 2.f * dt[3], 0.f));
                f[4] = sqrtf(fmaxf(nd.x + md.x - 2.f * dt[4], 0.f));
                f[5] = sqrtf(fmaxf(nd.y + md.y - 2.f * dt[5], 0.f));
                f[6] = sqrtf(fmaxf(nd.z + md.z - 2.f * dt[6], 0.f));
                f[7] = sqrtf(fmaxf(nd.w + md.w - 2.f * dt[7], 0.f));
            } else {                             // edges: streamed once -> non-temporal
                const int d = cb - 256;
                const f32x4 ea = __builtin_nontemporal_load((const f32x4*)(ed + d));
                const f32x4 eb = __builtin_nontemporal_load((const f32x4*)(ed + d + 4));
                f[0] = ea.x; f[1] = ea.y; f[2] = ea.z; f[3] = ea.w;
                f[4] = eb.x; f[5] = eb.y; f[6] = eb.z; f[7] = eb.w;
            }
#pragma unroll
            for (int j = 0; j < 8; ++j) afr[ks].u[g * 8 + j] = f2bf(f[j]);
        }
    }

    // ---- complete W2 staging before the WMMA loops ----
#if __has_builtin(__builtin_amdgcn_global_load_async_to_lds_b128)
    asm volatile("s_wait_asynccnt 0x0" ::: "memory");
#endif
    __syncthreads();

    // ---- output accumulators: 8 N-tiles of 16 columns ----
    v8f oacc[8];
#pragma unroll
    for (int t = 0; t < 8; ++t)
#pragma unroll
        for (int r = 0; r < 8; ++r) oacc[t][r] = 0.f;

    for (int ks2 = 0; ks2 < 8; ++ks2) {          // GEMM2 K-step = 2 GEMM1 N-tiles
        v8f ha, hb;
#pragma unroll
        for (int r = 0; r < 8; ++r) { ha[r] = 0.f; hb[r] = 0.f; }

        // GEMM1: 10 K-steps over feats; B fragments stream via vmem/L0
#pragma unroll
        for (int ks = 0; ks < 10; ++ks) {
            Frag16 bf0, bf1;
            const unsigned short* p0 = w1b + (ks2 * 32 + m) * IN_FEATS + 32 * ks + 8 * h;
            bf0.q[0] = *(const u32x4*)(p0);
            bf0.q[1] = *(const u32x4*)(p0 + 16);
            const unsigned short* p1 = p0 + 16 * IN_FEATS;
            bf1.q[0] = *(const u32x4*)(p1);
            bf1.q[1] = *(const u32x4*)(p1 + 16);
            ha = __builtin_amdgcn_wmma_f32_16x16x32_bf16(false, afr[ks].v, false, bf0.v,
                                                         (short)0, ha, false, false);
            hb = __builtin_amdgcn_wmma_f32_16x16x32_bf16(false, afr[ks].v, false, bf1.v,
                                                         (short)0, hb, false, false);
        }

        // bias + silu, write bf16 into per-wave transpose tile (C layout -> LDS)
        const float b1a = bias1[ks2 * 32 + m];
        const float b1b = bias1[ks2 * 32 + 16 + m];
#pragma unroll
        for (int r = 0; r < 8; ++r) {
            float xa = ha[r] + b1a;
            float xb = hb[r] + b1b;
            xa = xa / (1.0f + __expf(-xa));
            xb = xb / (1.0f + __expf(-xb));
            hT[wave][8 * h + r][m]      = f2bf(xa);
            hT[wave][8 * h + r][16 + m] = f2bf(xb);
        }
        asm volatile("s_wait_dscnt 0" ::: "memory");

        // read h back in A-fragment layout (lane = row m, contiguous K runs)
        Frag16 a2;
        a2.q[0] = *(const u32x4*)&hT[wave][m][8 * h];
        a2.q[1] = *(const u32x4*)&hT[wave][m][16 + 8 * h];

        // GEMM2: B fragments from LDS (padded rows -> conflict-free b128 reads)
#pragma unroll
        for (int ot = 0; ot < 8; ++ot) {
            Frag16 b2f;
            const unsigned short* p = &w2lds[ot * 16 + m][32 * ks2 + 8 * h];
            b2f.q[0] = *(const u32x4*)(p);
            b2f.q[1] = *(const u32x4*)(p + 16);
            oacc[ot] = __builtin_amdgcn_wmma_f32_16x16x32_bf16(false, a2.v, false, b2f.v,
                                                               (short)0, oacc[ot], false, false);
        }
    }

    // ---- bias2 + non-temporal f32 store of out [NQ, NK, 128] ----
#pragma unroll
    for (int ot = 0; ot < 8; ++ot) {
        const int col = ot * 16 + m;
        const float bv = bias2[col];
#pragma unroll
        for (int r = 0; r < 8; ++r) {
            const int kk = k0 + 8 * h + r;
            __builtin_nontemporal_store(oacc[ot][r] + bv,
                                        &out[(q * NK + kk) * D_OUT + col]);
        }
    }
}

// ---------------- launch ----------------

extern "C" void kernel_launch(void* const* d_in, const int* in_sizes, int n_in,
                              void* d_out, int out_size, void* d_ws, size_t ws_size,
                              hipStream_t stream) {
    (void)in_sizes; (void)n_in; (void)out_size; (void)ws_size;
    const float* q_equi = (const float*)d_in[0];
    const float* q_inv  = (const float*)d_in[1];
    const float* k_equi = (const float*)d_in[2];
    const float* k_inv  = (const float*)d_in[3];
    const float* edges  = (const float*)d_in[4];
    const float* qW     = (const float*)d_in[5];
    const float* qb     = (const float*)d_in[6];
    const float* kW     = (const float*)d_in[7];
    const float* kb     = (const float*)d_in[8];
    const float* W1     = (const float*)d_in[9];
    const float* b1     = (const float*)d_in[10];
    const float* W2     = (const float*)d_in[11];
    const float* b2     = (const float*)d_in[12];
    float* out = (float*)d_out;

    char* ws = (char*)d_ws;
    float* qmsg = (float*)ws;                  ws += (size_t)NQ * D_MSG * 4;
    float* kmsg = (float*)ws;                  ws += (size_t)NK * D_MSG * 4;
    float* qn2  = (float*)ws;                  ws += (size_t)NQ * D_EQUI * 4;
    float* kn2  = (float*)ws;                  ws += (size_t)NK * D_EQUI * 4;
    unsigned short* w1b = (unsigned short*)ws; ws += (size_t)D_FF * IN_FEATS * 2;
    unsigned short* w2b = (unsigned short*)ws; ws += (size_t)D_OUT * D_FF * 2;

    prep_msg<<<NQ + NK, D_MSG, 0, stream>>>(q_inv, k_inv, qW, qb, kW, kb, qmsg, kmsg);
    prep_norms<<<(2 * NQ * D_EQUI) / 256, 256, 0, stream>>>(q_equi, k_equi, qn2, kn2);
    prep_wcvt<<<(D_FF * IN_FEATS + D_OUT * D_FF + 255) / 256, 256, 0, stream>>>(W1, W2, w1b, w2b);
    pairwise_main<<<(NQ / 8) * (NK / 16), 256, 0, stream>>>(
        q_equi, k_equi, edges, b1, b2, qmsg, kmsg, qn2, kn2, w1b, w2b, out);
}
